// CaptionEncoderC_37108517438104
// MI455X (gfx1250) — matrix operand order repaired and checked by reading, not compiled, not run on hardware
//
#include <hip/hip_runtime.h>
#include <hip/hip_bf16.h>
#include <stdint.h>

// CDNA5 / gfx1250. wave32 only.
typedef __attribute__((ext_vector_type(16))) __bf16 v16bf;
typedef __attribute__((ext_vector_type(8)))  float  v8f;

#if defined(__HIP_DEVICE_COMPILE__) && \
    __has_builtin(__builtin_amdgcn_global_load_async_to_lds_b128) && \
    __has_builtin(__builtin_amdgcn_s_wait_asynccnt)
#define USE_ASYNC 1
typedef int i4_ __attribute__((vector_size(16)));
typedef __attribute__((address_space(1))) i4_* gp_i4;   // global int4*
typedef __attribute__((address_space(3))) i4_* lp_i4;   // LDS int4*
#else
#define USE_ASYNC 0
#endif

#define B_    256
#define T_    64
#define E_    512
#define H_    1024
#define G_    4096      // 4*H
#define KS_X  16        // E/32
#define KS_H  32        // H/32
#define KS_G  48        // (E+H)/32
#define KS_A  32        // H/32 (aff)
#define MT_   16        // B/16
#define CH    4         // k-steps per LDS chunk
#define NCH   12        // KS_G / CH

__device__ __forceinline__ float sigmoidf_(float x) {
  return 1.0f / (1.0f + __expf(-x));
}

#if USE_ASYNC
// Async global->LDS 16B copy (GLOBAL_LOAD_ASYNC_TO_LDS_B128, ASYNCcnt).
__device__ __forceinline__ void async_cp16(const __bf16* g, __bf16* l) {
  __builtin_amdgcn_global_load_async_to_lds_b128(
      (gp_i4)(uintptr_t)g, (lp_i4)(uint32_t)(uintptr_t)l, 0, 0);
}
#endif

// ---------------------------------------------------------------------------
// Init: zero h-fragment (A layout) + c state, build combined gate bias.
// ---------------------------------------------------------------------------
__global__ void lstm_init_kernel(__bf16* __restrict__ ah, float* __restrict__ c,
                                 float* __restrict__ bias,
                                 const float* __restrict__ x2h_b,
                                 const float* __restrict__ h2h_b) {
  int idx = blockIdx.x * blockDim.x + threadIdx.x;   // 262144 threads
  ah[idx] = (__bf16)0.0f;
  c[idx]  = 0.0f;
  if (idx < G_) bias[idx] = x2h_b[idx] + h2h_b[idx];
}

// ---------------------------------------------------------------------------
// Pack combined weight [x2h_w ; h2h_w] (1536 x 4096) into B-fragment layout:
// Bpack[nt][ks][lane][e], lane: n = lane&15, half = lane>>4, k = ks*32+(e<8 ? half*8+e : 16+half*8+(e-8))
// ---------------------------------------------------------------------------
__global__ void pack_gates_w_kernel(const float* __restrict__ x2h_w,
                                    const float* __restrict__ h2h_w,
                                    __bf16* __restrict__ bp) {
  int idx  = blockIdx.x * blockDim.x + threadIdx.x;  // 256*48*32*16 = 6291456
  int e    = idx & 15;
  int lane = (idx >> 4) & 31;
  int ks   = (idx >> 9) % KS_G;
  int nt   = (idx >> 9) / KS_G;
  int n = nt * 16 + (lane & 15);
  int k = ks * 32 + ((e < 8) ? ((lane >> 4) * 8 + e) : (16 + (lane >> 4) * 8 + (e - 8)));
  float v = (k < E_) ? x2h_w[(size_t)k * G_ + n]
                     : h2h_w[(size_t)(k - E_) * G_ + n];
  bp[idx] = (__bf16)v;
}

// ---------------------------------------------------------------------------
// Pack aff_w (1024 x 1024) into B-fragment layout.
// ---------------------------------------------------------------------------
__global__ void pack_aff_w_kernel(const float* __restrict__ aff_w,
                                  __bf16* __restrict__ bp) {
  int idx  = blockIdx.x * blockDim.x + threadIdx.x;  // 64*32*32*16 = 1048576
  int e    = idx & 15;
  int lane = (idx >> 4) & 31;
  int ks   = (idx >> 9) & 31;
  int nt   = idx >> 14;
  int n = nt * 16 + (lane & 15);
  int k = ks * 32 + ((e < 8) ? ((lane >> 4) * 8 + e) : (16 + (lane >> 4) * 8 + (e - 8)));
  bp[idx] = (__bf16)aff_w[(size_t)k * H_ + n];
}

// ---------------------------------------------------------------------------
// Gather embedding rows, pack as A fragments (16x32 bf16 layout, ISA 7.12.2):
// Apack[t][mt][ks][lane][e]; r = lane&15, half = lane>>4,
// k_local = e<8 ? half*8+e : 16+half*8+(e-8)
// ---------------------------------------------------------------------------
__global__ void pack_x_kernel(const int* __restrict__ seq,
                              const float* __restrict__ embed,
                              __bf16* __restrict__ ap) {
  int idx  = blockIdx.x * blockDim.x + threadIdx.x;  // 64*16*16*32*16 = 8388608
  int e    = idx & 15;
  int lane = (idx >> 4) & 31;
  int ks   = (idx >> 9) & 15;
  int mt   = (idx >> 13) & 15;
  int t    = idx >> 17;
  int r    = lane & 15;
  int half = lane >> 4;
  int kl   = (e < 8) ? (half * 8 + e) : (16 + half * 8 + (e - 8));
  int k    = ks * 32 + kl;
  int b    = mt * 16 + r;
  int tok  = seq[b * T_ + t];
  ap[idx]  = (__bf16)embed[(size_t)tok * E_ + k];
}

// ---------------------------------------------------------------------------
// B-stage address helper: flat 16B unit f (0..1023) of one LDS chunk
// (4 N-tiles x CH k-steps), maps to packed-global offset + LDS elem offset.
// ---------------------------------------------------------------------------
__device__ __forceinline__ void b_addr(int f, int ntq, int ks0,
                                       size_t* goff, int* loff) {
  int q   = f >> 8;        // N-tile within quad
  int rem = f & 255;
  int ksl = rem >> 6;      // k-step within chunk
  int c   = rem & 63;      // 16B unit within (tile, ks) slab
  *goff = ((size_t)(ntq * 4 + q) * KS_G + (size_t)(ks0 + ksl)) * 512 +
          (size_t)c * 8;
  *loff = f * 8;           // elems
}

// ---------------------------------------------------------------------------
// Per-step gate GEMM: gates(256x4096) = [x_t ; h](256x1536) @ Wcomb(1536x4096)
// Block = 512 threads = 16 waves = all 16 M-tiles; block owns 64 N columns.
// B fragments staged through double-buffered LDS (async-to-LDS when available)
// so each B byte is fetched from L2 once per block. Each wave: 16x64 strip,
// 4 x v_wmma_f32_16x16x32_bf16 per A fragment.
// ---------------------------------------------------------------------------
__global__ void __launch_bounds__(512)
gemm_gates_kernel(const __bf16* __restrict__ ax, const __bf16* __restrict__ ah,
                  const __bf16* __restrict__ bw, float* __restrict__ gates,
                  int t) {
  __shared__ __align__(32) __bf16 smem[2][4 * CH * 512];   // 2 x 16 KB
  const int tid  = threadIdx.x;
  const int lane = tid & 31;
  const int mt   = tid >> 5;          // wave id == M tile (wave32)
  const int ntq  = blockIdx.x;        // 0..63 quad of N tiles

  const __bf16* axb = ax + ((((size_t)t * MT_ + mt) * KS_X) * 32 + lane) * 16;
  const __bf16* ahb = ah + (((size_t)mt * KS_H) * 32 + lane) * 16;

  v8f acc[4] = {};

  // ---- prologue: stage chunk 0 into buffer 0 ----
  {
    size_t g0, g1; int l0, l1;
    b_addr(tid, ntq, 0, &g0, &l0);
    b_addr(tid + 512, ntq, 0, &g1, &l1);
#if USE_ASYNC
    async_cp16(bw + g0, &smem[0][l0]);
    async_cp16(bw + g1, &smem[0][l1]);
    __builtin_amdgcn_s_wait_asynccnt(0);
#else
    *(uint4*)&smem[0][l0] = *(const uint4*)(bw + g0);
    *(uint4*)&smem[0][l1] = *(const uint4*)(bw + g1);
#endif
  }
  __syncthreads();

  int cur = 0;
  for (int ck = 0; ck < NCH; ++ck) {
    const int nb = cur ^ 1;
    const bool has_next = (ck + 1 < NCH);

    // ---- issue next chunk's B loads before computing current chunk ----
#if USE_ASYNC
    if (has_next) {
      size_t g0, g1; int l0, l1;
      b_addr(tid, ntq, (ck + 1) * CH, &g0, &l0);
      b_addr(tid + 512, ntq, (ck + 1) * CH, &g1, &l1);
      async_cp16(bw + g0, &smem[nb][l0]);
      async_cp16(bw + g1, &smem[nb][l1]);
    }
#else
    uint4 r0, r1; int l0 = 0, l1 = 0;
    if (has_next) {
      size_t g0, g1;
      b_addr(tid, ntq, (ck + 1) * CH, &g0, &l0);
      b_addr(tid + 512, ntq, (ck + 1) * CH, &g1, &l1);
      r0 = *(const uint4*)(bw + g0);
      r1 = *(const uint4*)(bw + g1);
    }
#endif

    // ---- compute current chunk: no per-k branch, A source per chunk ----
    {
      const __bf16* abase = (ck < 4)
          ? (axb + (size_t)ck * (CH * 512))
          : (ahb + (size_t)(ck - 4) * (CH * 512));
#pragma unroll
      for (int ksl = 0; ksl < CH; ++ksl) {
        v16bf a = *(const v16bf*)(abase + (size_t)ksl * 512);
#pragma unroll
        for (int q = 0; q < 4; ++q) {
          v16bf b = *(const v16bf*)
              &smem[cur][(q * CH + ksl) * 512 + lane * 16];
          acc[q] = __builtin_amdgcn_wmma_f32_16x16x32_bf16(
              false, a, false, b, (short)0, acc[q], false, false);
        }
      }
    }

    // ---- land next chunk into the other buffer ----
#if USE_ASYNC
    if (has_next) __builtin_amdgcn_s_wait_asynccnt(0);
#else
    if (has_next) {
      *(uint4*)&smem[nb][l0] = r0;
      *(uint4*)&smem[nb][l1] = r1;
    }
#endif
    __syncthreads();
    cur = nb;
  }

  // C/D layout: VGPR i, lanes 0-15 -> M=i,N=lane; lanes 16-31 -> M=8+i
  const int m = mt * 16 + (lane >> 4) * 8;
  const int n = ntq * 64 + (lane & 15);
#pragma unroll
  for (int q = 0; q < 4; ++q)
#pragma unroll
    for (int i = 0; i < 8; ++i)
      gates[(size_t)(m + i) * G_ + n + q * 16] = acc[q][i];
}

// ---------------------------------------------------------------------------
// Per-step elementwise LSTM cell + masking; updates c (f32) and h directly
// in bf16 A-fragment layout for the next step's GEMM.
// ---------------------------------------------------------------------------
__global__ void lstm_cell_kernel(const float* __restrict__ gates,
                                 const float* __restrict__ bias,
                                 const int* __restrict__ seq_len,
                                 float* __restrict__ c,
                                 __bf16* __restrict__ ah,
                                 float* __restrict__ out_hs,
                                 float* __restrict__ out_cs, int t) {
  int idx = blockIdx.x * blockDim.x + threadIdx.x;   // 262144
  int b = idx >> 10;
  int j = idx & 1023;
  size_t gb = (size_t)b * G_ + j;
  float iv = gates[gb]          + bias[j];
  float fv = gates[gb + 1 * H_] + bias[j + 1 * H_];
  float gv = gates[gb + 2 * H_] + bias[j + 2 * H_];
  float ov = gates[gb + 3 * H_] + bias[j + 3 * H_];
  float c_old = c[idx];
  float c_new = sigmoidf_(fv) * c_old + sigmoidf_(iv) * tanhf(gv);
  float h_new = sigmoidf_(ov) * tanhf(c_new);
  bool valid = t < seq_len[b];

  size_t ob = ((size_t)b * T_ + t) * H_ + j;
  out_hs[ob] = valid ? h_new : 0.0f;
  out_cs[ob] = valid ? c_new : 0.0f;

  if (valid) {
    c[idx] = c_new;
    int mt = b >> 4, r = b & 15;
    int ks = j >> 5, kk = j & 31;
    int half, e;
    if      (kk < 8)  { half = 0; e = kk; }
    else if (kk < 16) { half = 1; e = kk - 8; }
    else if (kk < 24) { half = 0; e = kk - 8; }
    else              { half = 1; e = kk - 16; }
    int lane = half * 16 + r;
    ah[(((size_t)mt * KS_H + ks) * 32 + lane) * 16 + e] = (__bf16)h_new;
  }
}

// ---------------------------------------------------------------------------
// final_hidden = tanh(h_last @ aff_w + aff_b); 512 waves, WMMA bf16.
// ---------------------------------------------------------------------------
__global__ void gemm_final_kernel(const __bf16* __restrict__ ah,
                                  const __bf16* __restrict__ bw,
                                  const float* __restrict__ aff_b,
                                  float* __restrict__ outf) {
  int gtid = blockIdx.x * blockDim.x + threadIdx.x;
  int w    = gtid >> 5;
  int lane = threadIdx.x & 31;
  int mt   = w & 15;
  int ntp  = w >> 4;             // 32 N-tile pairs
  int n0t  = ntp * 2;

  v8f acc0 = {}, acc1 = {};
  const __bf16* ahb = ah + (((size_t)mt * KS_H) * 32 + lane) * 16;
  const __bf16* b0b = bw + ((((size_t)n0t)     * KS_A) * 32 + lane) * 16;
  const __bf16* b1b = bw + ((((size_t)n0t + 1) * KS_A) * 32 + lane) * 16;
#pragma unroll 2
  for (int ks = 0; ks < KS_A; ++ks) {
    v16bf a  = *(const v16bf*)(ahb + (size_t)ks * 512);
    v16bf b0 = *(const v16bf*)(b0b + (size_t)ks * 512);
    v16bf b1 = *(const v16bf*)(b1b + (size_t)ks * 512);
    acc0 = __builtin_amdgcn_wmma_f32_16x16x32_bf16(
        false, a, false, b0, (short)0, acc0, false, false);
    acc1 = __builtin_amdgcn_wmma_f32_16x16x32_bf16(
        false, a, false, b1, (short)0, acc1, false, false);
  }
  int m = mt * 16 + (lane >> 4) * 8;
  int n = n0t * 16 + (lane & 15);
#pragma unroll
  for (int i = 0; i < 8; ++i) {
    outf[(size_t)(m + i) * H_ + n]      = tanhf(acc0[i] + aff_b[n]);
    outf[(size_t)(m + i) * H_ + n + 16] = tanhf(acc1[i] + aff_b[n + 16]);
  }
}

__global__ void mask_kernel(const int* __restrict__ seq_len,
                            float* __restrict__ out_mask) {
  int idx = blockIdx.x * blockDim.x + threadIdx.x;   // 16384
  int b = idx / T_, t = idx % T_;
  out_mask[idx] = (t < seq_len[b]) ? 1.0f : 0.0f;
}

// ---------------------------------------------------------------------------
extern "C" void kernel_launch(void* const* d_in, const int* in_sizes, int n_in,
                              void* d_out, int out_size, void* d_ws,
                              size_t ws_size, hipStream_t stream) {
  const int*   seq     = (const int*)d_in[0];
  const int*   seq_len = (const int*)d_in[1];
  const float* embed   = (const float*)d_in[2];
  const float* x2h_w   = (const float*)d_in[3];
  const float* x2h_b   = (const float*)d_in[4];
  const float* h2h_w   = (const float*)d_in[5];
  const float* h2h_b   = (const float*)d_in[6];
  const float* aff_w   = (const float*)d_in[7];
  const float* aff_b   = (const float*)d_in[8];

  uint8_t* ws = (uint8_t*)d_ws;
  __bf16* bp_gates = (__bf16*)(ws);                         // 12,582,912 B
  __bf16* bp_aff   = (__bf16*)(ws + 12582912);              //  2,097,152 B
  __bf16* ap_x     = (__bf16*)(ws + 14680064);              // 16,777,216 B
  __bf16* ap_h     = (__bf16*)(ws + 31457280);              //    524,288 B
  float*  c_state  = (float*) (ws + 31981568);              //  1,048,576 B
  float*  gates    = (float*) (ws + 33030144);              //  4,194,304 B
  float*  bias     = (float*) (ws + 37224448);              //     16,384 B

  float* out_hs   = (float*)d_out;                          // (B,T,H)
  float* out_cs   = out_hs + (size_t)B_ * T_ * H_;          // (B,T,H)
  float* out_fin  = out_cs + (size_t)B_ * T_ * H_;          // (B,H)
  float* out_mask = out_fin + (size_t)B_ * H_;              // (B,T)

  lstm_init_kernel<<<1024, 256, 0, stream>>>(ap_h, c_state, bias, x2h_b, h2h_b);
  pack_gates_w_kernel<<<24576, 256, 0, stream>>>(x2h_w, h2h_w, bp_gates);
  pack_aff_w_kernel<<<4096, 256, 0, stream>>>(aff_w, bp_aff);
  pack_x_kernel<<<32768, 256, 0, stream>>>(seq, embed, ap_x);

  for (int t = 0; t < T_; ++t) {
    gemm_gates_kernel<<<64, 512, 0, stream>>>(ap_x, ap_h, bp_gates, gates, t);
    lstm_cell_kernel<<<1024, 256, 0, stream>>>(gates, bias, seq_len, c_state,
                                               ap_h, out_hs, out_cs, t);
  }

  gemm_final_kernel<<<64, 256, 0, stream>>>(ap_h, bp_aff, aff_b, out_fin);
  mask_kernel<<<64, 256, 0, stream>>>(seq_len, out_mask);
}
